// RSSMCell_5935644803078
// MI455X (gfx1250) — matrix operand order repaired and checked by reading, not compiled, not run on hardware
//
#include <hip/hip_runtime.h>
#include <hip/hip_bf16.h>

// ---------------------------------------------------------------------------
// Types for gfx1250 WMMA / async-LDS
// ---------------------------------------------------------------------------
typedef __attribute__((ext_vector_type(16))) __bf16 v16bf;
typedef __attribute__((ext_vector_type(8)))  __bf16 v8bf;
typedef __attribute__((ext_vector_type(8)))  float  v8f;
typedef __attribute__((ext_vector_type(4)))  int    v4i;
// builtin prototype (from hipcc diagnostic): param0 = global (AS1) v4i*
typedef __attribute__((address_space(1))) v4i* gas_v4i_ptr;
typedef __attribute__((address_space(3))) v4i* las_v4i_ptr;

#if defined(__AMDGCN__) && __has_builtin(__builtin_amdgcn_global_load_async_to_lds_b128)
#define USE_ASYNC_LDS 1
#endif

__device__ __forceinline__ unsigned short f32_to_bf16(float f) {
    unsigned u = __builtin_bit_cast(unsigned, f);
    u += 0x7fffu + ((u >> 16) & 1u);          // round-to-nearest-even
    return (unsigned short)(u >> 16);
}

__device__ __forceinline__ void wait_async0() {
#if defined(__AMDGCN__)
#if __has_builtin(__builtin_amdgcn_s_wait_asynccnt)
    __builtin_amdgcn_s_wait_asynccnt(0);
#else
    asm volatile("s_wait_asynccnt 0" ::: "memory");
#endif
#endif
}

// ---------------------------------------------------------------------------
// Weight prep: f32 [K x N] -> bf16 [N x K]  (transpose so B-fragments of the
// WMMA are contiguous 16-element runs along K). LDS-tiled for coalescing.
// K, N are multiples of 32.
// ---------------------------------------------------------------------------
__global__ __launch_bounds__(256) void transpose_cvt(
    const float* __restrict__ W, unsigned short* __restrict__ Wt, int K, int N)
{
    __shared__ float tile[32][33];
    const int tx = threadIdx.x & 31;
    const int ty = threadIdx.x >> 5;              // 0..7
    const int n0 = blockIdx.x * 32;
    const int k0 = blockIdx.y * 32;
#pragma unroll
    for (int i = 0; i < 32; i += 8)
        tile[ty + i][tx] = W[(size_t)(k0 + ty + i) * N + (n0 + tx)];
    __syncthreads();
#pragma unroll
    for (int i = 0; i < 32; i += 8)
        Wt[(size_t)(n0 + ty + i) * K + (k0 + tx)] = f32_to_bf16(tile[tx][ty + i]);
}

// f32 [rows x cols] -> bf16 packed into wider row (for concat inputs)
__global__ __launch_bounds__(256) void cvt_pack_bf16(
    const float* __restrict__ in, unsigned short* __restrict__ out,
    int rows, int cols, int ldOut, int colOff)
{
    int idx = blockIdx.x * blockDim.x + threadIdx.x;
    if (idx >= rows * cols) return;
    int r = idx / cols, c = idx % cols;
    out[(size_t)r * ldOut + colOff + c] = f32_to_bf16(in[idx]);
}

// ---------------------------------------------------------------------------
// bf16 WMMA GEMM:  C[M,N] = epilogue(A[M,K] @ Bt[N,K]^T + bias[N])
// 256 threads = 8 wave32 (4 M-waves x 2 N-waves); block tile 128x128; K step 32.
// Double-buffered async global->LDS staging (ASYNCcnt) when available;
// 8 v_wmma_f32_16x16x32_bf16 per K iteration per wave.
// ---------------------------------------------------------------------------
enum { EPI_F32_BIAS = 0, EPI_RELU_BF16 = 1, EPI_RELU_BF16_SHIFT = 2, EPI_SOFTPLUS_F32 = 3 };

template <int EPI>
__global__ __launch_bounds__(256) void gemm_bf16_wmma(
    const unsigned short* __restrict__ A,   // M x K   bf16 row-major
    const unsigned short* __restrict__ Bt,  // N x K   bf16 row-major (pre-transposed)
    const float* __restrict__ bias,         // N
    float* __restrict__ Cf, int ldC,        // f32 out (EPI_F32_BIAS / EPI_SOFTPLUS_F32)
    unsigned short* __restrict__ Cb0,       // bf16 out, ld = N
    unsigned short* __restrict__ Cb1,       // bf16 out (x+0.54), ld = N
    int M, int N, int K)
{
    (void)M;
    const int tid   = threadIdx.x;
    const int wave  = tid >> 5;
    const int lane  = tid & 31;
    const int waveM = wave >> 1;          // 0..3
    const int waveN = wave & 1;           // 0..1
    const int r     = lane & 15;          // row-in-tile (A) / col-in-tile (B,C)
    const int kh    = lane >> 4;          // 0/1 half-selector

    const int rowBlk = blockIdx.y * 128;
    const int colBlk = blockIdx.x * 128;

    const v8f zero = {};
    v8f acc[2][4];
#pragma unroll
    for (int mi = 0; mi < 2; ++mi)
#pragma unroll
        for (int ni = 0; ni < 4; ++ni) acc[mi][ni] = zero;

    // cooperative staging: each thread moves one 32B (16 bf16) segment per slab
    const int ldRow = tid >> 1;            // 0..127
    const int ldSeg = (tid & 1) * 16;      // 0 / 16

#ifdef USE_ASYNC_LDS
    // -------- double-buffered async global->LDS staging (one barrier/iter) ---
    __shared__ __attribute__((aligned(32))) unsigned short lA[2][128 * 48]; // 96B rows
    __shared__ __attribute__((aligned(32))) unsigned short lB[2][128 * 48];

    auto issueSlab = [&](int kk, int buf) {
        unsigned short* srcA = (unsigned short*)(A  + (size_t)(rowBlk + ldRow) * K + kk + ldSeg);
        unsigned short* srcB = (unsigned short*)(Bt + (size_t)(colBlk + ldRow) * K + kk + ldSeg);
        unsigned short* dstA = &lA[buf][ldRow * 48 + ldSeg];
        unsigned short* dstB = &lB[buf][ldRow * 48 + ldSeg];
        // imm offset is added to BOTH global and LDS address (ISA 10.x) -> the
        // two 16B halves of the 32B segment are contiguous in both spaces.
        __builtin_amdgcn_global_load_async_to_lds_b128(
            (gas_v4i_ptr)srcA, (las_v4i_ptr)dstA, 0, 0);
        __builtin_amdgcn_global_load_async_to_lds_b128(
            (gas_v4i_ptr)srcA, (las_v4i_ptr)dstA, 16, 0);
        __builtin_amdgcn_global_load_async_to_lds_b128(
            (gas_v4i_ptr)srcB, (las_v4i_ptr)dstB, 0, 0);
        __builtin_amdgcn_global_load_async_to_lds_b128(
            (gas_v4i_ptr)srcB, (las_v4i_ptr)dstB, 16, 0);
    };

    issueSlab(0, 0);
    int buf = 0;
    for (int kk = 0; kk < K; kk += 32) {
        wait_async0();        // my slab transfers done
        __syncthreads();      // everyone's done -> slab[buf] readable; slab[buf^1] free
        if (kk + 32 < K) issueSlab(kk + 32, buf ^ 1); // overlap next fill with WMMAs

        union { v16bf v; v8bf h[2]; } afrag[2];
#pragma unroll
        for (int mi = 0; mi < 2; ++mi) {
            const unsigned short* ar = &lA[buf][(waveM * 32 + mi * 16 + r) * 48];
            afrag[mi].v = *(const v16bf*)(ar + kh * 16);
        }
        v16bf bfrag[4];
#pragma unroll
        for (int ni = 0; ni < 4; ++ni)
            bfrag[ni] = *(const v16bf*)(&lB[buf][(waveN * 64 + ni * 16 + r) * 48 + kh * 16]);

#pragma unroll
        for (int mi = 0; mi < 2; ++mi)
#pragma unroll
            for (int ni = 0; ni < 4; ++ni)
                acc[mi][ni] = __builtin_amdgcn_wmma_f32_16x16x32_bf16(
                    false, afrag[mi].v, false, bfrag[ni],
                    (short)0, acc[mi][ni], false, false);
        buf ^= 1;
    }
#else
    // -------- fallback: synchronous staging through VGPRs --------------------
    __shared__ __attribute__((aligned(32))) unsigned short lA[128 * 48];
    __shared__ __attribute__((aligned(32))) unsigned short lB[128 * 48];

    for (int kk = 0; kk < K; kk += 32) {
        const unsigned short* srcA = A  + (size_t)(rowBlk + ldRow) * K + kk + ldSeg;
        const unsigned short* srcB = Bt + (size_t)(colBlk + ldRow) * K + kk + ldSeg;
        *(v16bf*)(&lA[ldRow * 48 + ldSeg]) = *(const v16bf*)srcA;
        *(v16bf*)(&lB[ldRow * 48 + ldSeg]) = *(const v16bf*)srcB;
        if (kk + 32 < K) {
            __builtin_prefetch(srcA + 32, 0, 1);
            __builtin_prefetch(srcB + 32, 0, 1);
        }
        __syncthreads();

        union { v16bf v; v8bf h[2]; } afrag[2];
#pragma unroll
        for (int mi = 0; mi < 2; ++mi) {
            const unsigned short* ar = &lA[(waveM * 32 + mi * 16 + r) * 48];
            afrag[mi].v = *(const v16bf*)(ar + kh * 16);
        }
        v16bf bfrag[4];
#pragma unroll
        for (int ni = 0; ni < 4; ++ni)
            bfrag[ni] = *(const v16bf*)(&lB[(waveN * 64 + ni * 16 + r) * 48 + kh * 16]);

#pragma unroll
        for (int mi = 0; mi < 2; ++mi)
#pragma unroll
            for (int ni = 0; ni < 4; ++ni)
                acc[mi][ni] = __builtin_amdgcn_wmma_f32_16x16x32_bf16(
                    false, afrag[mi].v, false, bfrag[ni],
                    (short)0, acc[mi][ni], false, false);
        __syncthreads();
    }
#endif

    // epilogue — C layout: VGPR e holds (row = e + 8*kh, col = r) of each 16x16 tile
#pragma unroll
    for (int mi = 0; mi < 2; ++mi) {
#pragma unroll
        for (int ni = 0; ni < 4; ++ni) {
            const int col  = colBlk + waveN * 64 + ni * 16 + r;
            const float bc = bias[col];
#pragma unroll
            for (int e = 0; e < 8; ++e) {
                const int row = rowBlk + waveM * 32 + mi * 16 + e + kh * 8;
                float v = acc[mi][ni][e] + bc;
                if (EPI == EPI_F32_BIAS) {
                    Cf[(size_t)row * ldC + col] = v;
                } else if (EPI == EPI_SOFTPLUS_F32) {
                    float sp = (v > 20.f) ? v : log1pf(__expf(v));
                    Cf[(size_t)row * ldC + col] = sp + 1e-4f;
                } else {
                    v = fmaxf(v, 0.f);
                    Cb0[(size_t)row * N + col] = f32_to_bf16(v);
                    if (EPI == EPI_RELU_BF16_SHIFT)
                        Cb1[(size_t)row * N + col] = f32_to_bf16(v + 0.54f);
                }
            }
        }
    }
}

// ---------------------------------------------------------------------------
// GRU gate fusion: gi/gh are [B x 3D] f32 (bias already added by GEMM).
// det = z*h + (1-z)*n ; optionally writes f32 (strided, into d_out) and up to
// two bf16 copies (plain, and packed into the [det|obs] concat buffer).
// ---------------------------------------------------------------------------
__global__ __launch_bounds__(256) void gru_combine(
    const float* __restrict__ gi, const float* __restrict__ gh,
    const float* __restrict__ h,
    float* __restrict__ outF, int ldF,
    unsigned short* __restrict__ outB0,
    unsigned short* __restrict__ outB1, int ldB1,
    int Bn, int Dd)
{
    int idx = blockIdx.x * blockDim.x + threadIdx.x;
    if (idx >= Bn * Dd) return;
    int b = idx / Dd, d = idx % Dd;
    size_t base = (size_t)b * 3 * Dd;
    float iz = gi[base + d], ir = gi[base + Dd + d], ih = gi[base + 2 * Dd + d];
    float hz = gh[base + d], hr = gh[base + Dd + d], hh = gh[base + 2 * Dd + d];
    float z  = 1.f / (1.f + __expf(-(iz + hz)));
    float rr = 1.f / (1.f + __expf(-(ir + hr)));
    float n  = tanhf(ih + rr * hh);
    float det = z * h[(size_t)b * Dd + d] + (1.f - z) * n;
    if (outF) outF[(size_t)b * ldF + d] = det;
    unsigned short bf = f32_to_bf16(det);
    outB0[(size_t)b * Dd + d] = bf;
    if (outB1) outB1[(size_t)b * ldB1 + d] = bf;
}

// sample = mu + std * eps, reading mu/std slices already in d_out
__global__ __launch_bounds__(256) void sample_fuse(
    float* __restrict__ out, const float* __restrict__ eps,
    int offMu, int offStd, int offS, int Bn, int Sn, int ldOut)
{
    int idx = blockIdx.x * blockDim.x + threadIdx.x;
    if (idx >= Bn * Sn) return;
    int b = idx / Sn, j = idx % Sn;
    size_t rb = (size_t)b * ldOut;
    out[rb + offS + j] = out[rb + offMu + j] + out[rb + offStd + j] * eps[idx];
}

// ---------------------------------------------------------------------------
// Host orchestration
// ---------------------------------------------------------------------------
extern "C" void kernel_launch(void* const* d_in, const int* in_sizes, int n_in,
                              void* d_out, int out_size, void* d_ws, size_t ws_size,
                              hipStream_t stream)
{
    (void)in_sizes; (void)n_in; (void)out_size; (void)ws_size;
    constexpr int Bn = 1024, S = 1024, D = 2048, E = 2048, CTX = 1024, OBS = 2048;
    constexpr int LDO = 1024 + 1024 + 1024 + 2048 + 1024 + 1024 + 1024; // 8192

    const float* obs      = (const float*)d_in[0];
    const float* context  = (const float*)d_in[1];
    const float* prev_s   = (const float*)d_in[2];
    const float* prev_det = (const float*)d_in[3];
    /* d_in[4] reset_state: forced to ones in reference -> ignored */
    const float* eps_pr   = (const float*)d_in[5];
    const float* eps_po   = (const float*)d_in[6];
    const float* W_p1 = (const float*)d_in[7];  const float* b_p1 = (const float*)d_in[8];
    const float* Wi   = (const float*)d_in[9];  const float* Wh   = (const float*)d_in[10];
    const float* bi   = (const float*)d_in[11]; const float* bh   = (const float*)d_in[12];
    const float* W_p2 = (const float*)d_in[13]; const float* b_p2 = (const float*)d_in[14];
    const float* W_pmu  = (const float*)d_in[15]; const float* b_pmu  = (const float*)d_in[16];
    const float* W_pstd = (const float*)d_in[17]; const float* b_pstd = (const float*)d_in[18];
    const float* W_q1 = (const float*)d_in[19]; const float* b_q1 = (const float*)d_in[20];
    const float* W_q2 = (const float*)d_in[21]; const float* b_q2 = (const float*)d_in[22];
    const float* W_qmu  = (const float*)d_in[23]; const float* b_qmu  = (const float*)d_in[24];
    const float* W_qstd = (const float*)d_in[25]; const float* b_qstd = (const float*)d_in[26];
    float* out = (float*)d_out;

    // carve workspace
    char* cur = (char*)d_ws;
    auto take = [&](size_t bytes) -> void* {
        void* p = (void*)cur; cur += (bytes + 255) & ~(size_t)255; return p;
    };
    typedef unsigned short bfr;
    bfr* Wp1t  = (bfr*)take((size_t)(S+CTX)*E*2);     // [E x 2048]
    bfr* Wit   = (bfr*)take((size_t)E*3*D*2);         // [6144 x 2048]
    bfr* Wht   = (bfr*)take((size_t)D*3*D*2);         // [6144 x 2048]
    bfr* Wp2t  = (bfr*)take((size_t)D*E*2);
    bfr* Wpmut = (bfr*)take((size_t)E*S*2);
    bfr* Wpstdt= (bfr*)take((size_t)E*S*2);
    bfr* Wq1t  = (bfr*)take((size_t)(D+OBS)*E*2);     // [2048 x 4096]
    bfr* Wq2t  = (bfr*)take((size_t)D*E*2);
    bfr* Wqmut = (bfr*)take((size_t)E*S*2);
    bfr* Wqstdt= (bfr*)take((size_t)E*S*2);
    bfr* Ap1   = (bfr*)take((size_t)Bn*(S+CTX)*2);    // [prev_sample | context]
    bfr* Hbf   = (bfr*)take((size_t)Bn*D*2);          // prev_det_state bf16
    bfr* H1    = (bfr*)take((size_t)Bn*E*2);
    bfr* Det   = (bfr*)take((size_t)Bn*D*2);
    bfr* Aq1   = (bfr*)take((size_t)Bn*(D+OBS)*2);    // [det | obs]
    bfr* H2    = (bfr*)take((size_t)Bn*E*2);
    bfr* H2s   = (bfr*)take((size_t)Bn*E*2);
    bfr* H1q   = (bfr*)take((size_t)Bn*E*2);
    bfr* Detq  = (bfr*)take((size_t)Bn*D*2);
    bfr* H2q   = (bfr*)take((size_t)Bn*E*2);
    bfr* H2qs  = (bfr*)take((size_t)Bn*E*2);
    float* gi  = (float*)take((size_t)Bn*3*D*4);
    float* gh  = (float*)take((size_t)Bn*3*D*4);

    const dim3 blk(256);
    auto tgrid = [](int K, int N) { return dim3(N / 32, K / 32); };
    auto ggrid = [](int M, int N) { return dim3(N / 128, M / 128); };
    auto egrid = [](int n) { return dim3((n + 255) / 256); };

    // --- 1. weight prep: f32 [KxN] -> bf16 [NxK] (bf16 set fits in 192MB L2)
    transpose_cvt<<<tgrid(S+CTX, E), blk, 0, stream>>>(W_p1,  Wp1t,  S+CTX, E);
    transpose_cvt<<<tgrid(E, 3*D),   blk, 0, stream>>>(Wi,    Wit,   E, 3*D);
    transpose_cvt<<<tgrid(D, 3*D),   blk, 0, stream>>>(Wh,    Wht,   D, 3*D);
    transpose_cvt<<<tgrid(D, E),     blk, 0, stream>>>(W_p2,  Wp2t,  D, E);
    transpose_cvt<<<tgrid(E, S),     blk, 0, stream>>>(W_pmu, Wpmut, E, S);
    transpose_cvt<<<tgrid(E, S),     blk, 0, stream>>>(W_pstd,Wpstdt,E, S);
    transpose_cvt<<<tgrid(D+OBS, E), blk, 0, stream>>>(W_q1,  Wq1t,  D+OBS, E);
    transpose_cvt<<<tgrid(D, E),     blk, 0, stream>>>(W_q2,  Wq2t,  D, E);
    transpose_cvt<<<tgrid(E, S),     blk, 0, stream>>>(W_qmu, Wqmut, E, S);
    transpose_cvt<<<tgrid(E, S),     blk, 0, stream>>>(W_qstd,Wqstdt,E, S);

    // --- 2. activation inputs -> bf16 (with concat packing)
    cvt_pack_bf16<<<egrid(Bn*S),   blk, 0, stream>>>(prev_s,   Ap1, Bn, S,   S+CTX, 0);
    cvt_pack_bf16<<<egrid(Bn*CTX), blk, 0, stream>>>(context,  Ap1, Bn, CTX, S+CTX, S);
    cvt_pack_bf16<<<egrid(Bn*D),   blk, 0, stream>>>(prev_det, Hbf, Bn, D,   D,     0);
    cvt_pack_bf16<<<egrid(Bn*OBS), blk, 0, stream>>>(obs,      Aq1, Bn, OBS, D+OBS, D);

    // --- 3. prior path
    gemm_bf16_wmma<EPI_RELU_BF16><<<ggrid(Bn, E), blk, 0, stream>>>(
        Ap1, Wp1t, b_p1, nullptr, 0, H1, nullptr, Bn, E, S+CTX);
    gemm_bf16_wmma<EPI_F32_BIAS><<<ggrid(Bn, 3*D), blk, 0, stream>>>(
        H1, Wit, bi, gi, 3*D, nullptr, nullptr, Bn, 3*D, E);
    gemm_bf16_wmma<EPI_F32_BIAS><<<ggrid(Bn, 3*D), blk, 0, stream>>>(
        Hbf, Wht, bh, gh, 3*D, nullptr, nullptr, Bn, 3*D, D);     // shared by both GRUs
    gru_combine<<<egrid(Bn*D), blk, 0, stream>>>(
        gi, gh, prev_det, out + 3072, LDO, Det, Aq1, D+OBS, Bn, D); // det -> out + concat buf
    gemm_bf16_wmma<EPI_RELU_BF16_SHIFT><<<ggrid(Bn, E), blk, 0, stream>>>(
        Det, Wp2t, b_p2, nullptr, 0, H2, H2s, Bn, E, D);
    gemm_bf16_wmma<EPI_F32_BIAS><<<ggrid(Bn, S), blk, 0, stream>>>(
        H2, Wpmut, b_pmu, out + 0, LDO, nullptr, nullptr, Bn, S, E);
    gemm_bf16_wmma<EPI_SOFTPLUS_F32><<<ggrid(Bn, S), blk, 0, stream>>>(
        H2s, Wpstdt, b_pstd, out + 1024, LDO, nullptr, nullptr, Bn, S, E);
    sample_fuse<<<egrid(Bn*S), blk, 0, stream>>>(out, eps_pr, 0, 1024, 2048, Bn, S, LDO);

    // --- 4. posterior path (gh reused; det already packed into Aq1 left half)
    gemm_bf16_wmma<EPI_RELU_BF16><<<ggrid(Bn, E), blk, 0, stream>>>(
        Aq1, Wq1t, b_q1, nullptr, 0, H1q, nullptr, Bn, E, D+OBS);
    gemm_bf16_wmma<EPI_F32_BIAS><<<ggrid(Bn, 3*D), blk, 0, stream>>>(
        H1q, Wit, bi, gi, 3*D, nullptr, nullptr, Bn, 3*D, E);
    gru_combine<<<egrid(Bn*D), blk, 0, stream>>>(
        gi, gh, prev_det, nullptr, 0, Detq, nullptr, 0, Bn, D);
    gemm_bf16_wmma<EPI_RELU_BF16_SHIFT><<<ggrid(Bn, E), blk, 0, stream>>>(
        Detq, Wq2t, b_q2, nullptr, 0, H2q, H2qs, Bn, E, D);
    gemm_bf16_wmma<EPI_F32_BIAS><<<ggrid(Bn, S), blk, 0, stream>>>(
        H2q, Wqmut, b_qmu, out + 5120, LDO, nullptr, nullptr, Bn, S, E);
    gemm_bf16_wmma<EPI_SOFTPLUS_F32><<<ggrid(Bn, S), blk, 0, stream>>>(
        H2qs, Wqstdt, b_qstd, out + 6144, LDO, nullptr, nullptr, Bn, S, E);
    sample_fuse<<<egrid(Bn*S), blk, 0, stream>>>(out, eps_po, 5120, 6144, 7168, Bn, S, LDO);
}
